// Hippocampus_47665547051395
// MI455X (gfx1250) — matrix operand order repaired and checked by reading, not compiled, not run on hardware
//
#include <hip/hip_runtime.h>
#include <math.h>

// ---------------------------------------------------------------------------
// Problem constants (from reference): B=4096, K=512, P=256, NP=64, N=32768
// DM = P + 3*NP = 448, hidden = 1024, combined = 768, out row = 449
// ---------------------------------------------------------------------------
#define BB   4096
#define KK   512
#define PP   256
#define NN   32768
#define DM   448
#define HID  1024
#define CMB  768
#define OUTW 449

typedef __bf16 bf16_t;
typedef bf16_t v16bf __attribute__((ext_vector_type(16)));
typedef float  v8f   __attribute__((ext_vector_type(8)));

// ---------------------------------------------------------------------------
// CDNA5 async global->LDS copy (GLOBAL_LOAD_ASYNC_TO_LDS_B128, ASYNCcnt)
// Builtin signature (from hipcc diagnostic): arg0 = int4 AS1* (global src),
// arg1 = LDS dst, arg2/3 = imm offset / cpol.
// ---------------------------------------------------------------------------
#if defined(__has_builtin)
#  if __has_builtin(__builtin_amdgcn_global_load_async_to_lds_b128)
#    define HAS_ASYNC_LDS 1
#  endif
#endif
#ifndef HAS_ASYNC_LDS
#  define HAS_ASYNC_LDS 0
#endif

typedef int v4i __attribute__((vector_size(16)));
typedef __attribute__((address_space(1))) v4i as1_v4i;
typedef __attribute__((address_space(3))) v4i as3_v4i;

__device__ __forceinline__ void async_copy16(bf16_t* lds, const bf16_t* glob) {
#if HAS_ASYNC_LDS
    __builtin_amdgcn_global_load_async_to_lds_b128(
        (as1_v4i*)glob, (as3_v4i*)lds, 0, 0);
#else
    *(uint4*)lds = *(const uint4*)glob;   // sync fallback: load + ds_store
#endif
}
__device__ __forceinline__ void wait_async_prev() {   // <=4 outstanding: prev tile done
#if HAS_ASYNC_LDS
    asm volatile("s_wait_asynccnt 0x4" ::: "memory");
#endif
}
__device__ __forceinline__ void wait_async_all() {
#if HAS_ASYNC_LDS
    asm volatile("s_wait_asynccnt 0x0" ::: "memory");
#endif
}

// ---------------------------------------------------------------------------
// Helpers
// ---------------------------------------------------------------------------
__device__ __forceinline__ unsigned ordered_bits(float f) {
    unsigned b = __float_as_uint(f);
    return (b & 0x80000000u) ? ~b : (b | 0x80000000u);
}
__device__ __forceinline__ float ordered_decode(unsigned ob) {
    unsigned b = (ob & 0x80000000u) ? (ob & 0x7FFFFFFFu) : ~ob;
    return __uint_as_float(b);
}
__device__ __forceinline__ unsigned long long shflxor_u64(unsigned long long x, int m) {
    unsigned lo = (unsigned)(x & 0xFFFFFFFFull);
    unsigned hi = (unsigned)(x >> 32);
    lo = __shfl_xor(lo, m, 32);
    hi = __shfl_xor(hi, m, 32);
    return ((unsigned long long)hi << 32) | (unsigned long long)lo;
}

// ---------------------------------------------------------------------------
// Elementwise kernels
// ---------------------------------------------------------------------------
__global__ void k_f32_to_bf16(const float* __restrict__ in, bf16_t* __restrict__ out, int n) {
    int i = blockIdx.x * blockDim.x + threadIdx.x;
    if (i < n) out[i] = (bf16_t)in[i];
}

__global__ void k_build_combined(const float* __restrict__ act, const float* __restrict__ pfc,
                                 bf16_t* __restrict__ out) {
    int i = blockIdx.x * blockDim.x + threadIdx.x;
    if (i >= BB * CMB) return;
    int b = i / CMB, c = i % CMB;
    float v = (c < KK) ? act[(size_t)b * KK + c] : pfc[(size_t)b * PP + (c - KK)];
    out[i] = (bf16_t)v;
}

__global__ void k_normalize_rows(const float* __restrict__ in, bf16_t* __restrict__ out, int cols) {
    __shared__ float red[256];
    int row = blockIdx.x, t = threadIdx.x;
    const float* r = in + (size_t)row * cols;
    float ss = 0.f;
    for (int c = t; c < cols; c += 256) { float x = r[c]; ss += x * x; }
    red[t] = ss;
    __syncthreads();
    for (int s = 128; s > 0; s >>= 1) {
        if (t < s) red[t] += red[t + s];
        __syncthreads();
    }
    float inv = 1.0f / fmaxf(sqrtf(red[0]), 1e-12f);
    bf16_t* o = out + (size_t)row * cols;
    for (int c = t; c < cols; c += 256) o[c] = (bf16_t)(r[c] * inv);
}

__global__ void k_zero_u64(unsigned long long* __restrict__ p, int n) {
    int i = blockIdx.x * blockDim.x + threadIdx.x;
    if (i < n) p[i] = 0ull;
}

// ---------------------------------------------------------------------------
// WMMA GEMM:  C[m,n] = sum_k A[m,k] * Bm[n,k]   (bf16 row-major, f32 acc)
// Block tile 128(M) x 128(N), 8 waves. Wave (mw = w>>2, nw = w&3) computes a
// 64(M) x 32(N) sub-tile: 4 A-frags x 2 B-frags -> 8 WMMAs per 32-wide K-step.
// Double-buffered LDS tiles filled with async global->LDS copies that overlap
// the WMMA work on the other buffer.
// EPI 0: relu(C + bias) -> bf16   EPI 1: (C + bias) -> f32
// EPI 2: fused per-row max/argmax -> packed u64 atomics
// ---------------------------------------------------------------------------
#define TILE_K 32
#define LDT    40   // padded LDS row stride (80 B, 16B-aligned chunks)

__device__ __forceinline__ void stage_tile(bf16_t* dstA, bf16_t* dstB,
                                           const bf16_t* __restrict__ A,
                                           const bf16_t* __restrict__ Bm,
                                           int m0, int n0, int k0, int Kdim,
                                           int r0, int lc) {
#pragma unroll
    for (int rr = 0; rr < 128; rr += 64) {
        async_copy16(dstA + (r0 + rr) * LDT + lc * 8,
                     A + (size_t)(m0 + r0 + rr) * Kdim + k0 + lc * 8);
        async_copy16(dstB + (r0 + rr) * LDT + lc * 8,
                     Bm + (size_t)(n0 + r0 + rr) * Kdim + k0 + lc * 8);
    }
}

template <int EPI>
__global__ void __launch_bounds__(256)
k_gemm_wmma(const bf16_t* __restrict__ A, const bf16_t* __restrict__ Bm,
            const float* __restrict__ bias,
            float* __restrict__ outF, bf16_t* __restrict__ outB,
            unsigned long long* __restrict__ packed,
            int Kdim, int Ncols)
{
    __shared__ __align__(16) bf16_t As[2][128 * LDT];
    __shared__ __align__(16) bf16_t Bs[2][128 * LDT];

    const int t    = threadIdx.x;
    const int wave = t >> 5;
    const int lane = t & 31;
    const int half = lane >> 4;
    const int r    = lane & 15;
    const int mw   = wave >> 2;       // 0..1
    const int nw   = wave & 3;        // 0..3
    const int m0   = blockIdx.y * 128;
    const int n0   = blockIdx.x * 128;
    const int r0   = t >> 2;          // 0..63 staging row
    const int lc   = t & 3;           // 16B chunk in 32-elem K slice

    v8f acc[4][2];
#pragma unroll
    for (int ms = 0; ms < 4; ++ms)
#pragma unroll
        for (int n = 0; n < 2; ++n)
#pragma unroll
            for (int v = 0; v < 8; ++v) acc[ms][n][v] = 0.f;

    const int nt = Kdim / TILE_K;
    stage_tile(As[0], Bs[0], A, Bm, m0, n0, 0, Kdim, r0, lc);

    for (int i = 0; i < nt; ++i) {
        const int c = i & 1;
        const bf16_t* curA = As[c];
        const bf16_t* curB = Bs[c];
        if (i + 1 < nt) {
            // prefetch next K tile into the other buffer (overlaps WMMA below)
            stage_tile(As[1 - c], Bs[1 - c], A, Bm, m0, n0, (i + 1) * TILE_K,
                       Kdim, r0, lc);
            wait_async_prev();     // only tile i must have landed
        } else {
            wait_async_all();
        }
        __syncthreads();           // everyone's tile i visible in LDS

        // ---- B fragments (32x16): lane r = col, contiguous K chunk @16*half
        union { uint4 u[2]; v16bf v; } fb[2];
#pragma unroll
        for (int n = 0; n < 2; ++n) {
            const bf16_t* pb = curB + (nw * 32 + n * 16 + r) * LDT + 16 * half;
            fb[n].u[0] = *(const uint4*)pb;
            fb[n].u[1] = *(const uint4*)(pb + 8);
        }
        // ---- A fragments (16x32): lane r = row, contiguous K chunk @16*half
#pragma unroll
        for (int ms = 0; ms < 4; ++ms) {
            union { uint4 u[2]; v16bf v; } fa;
            const bf16_t* pa = curA + (mw * 64 + ms * 16 + r) * LDT + 16 * half;
            fa.u[0] = *(const uint4*)pa;
            fa.u[1] = *(const uint4*)(pa + 8);
#pragma unroll
            for (int n = 0; n < 2; ++n)
                acc[ms][n] = __builtin_amdgcn_wmma_f32_16x16x32_bf16(
                    false, fa.v, false, fb[n].v, (short)0, acc[ms][n], false, false);
        }
        __syncthreads();           // all reads of buf c done before it is refilled
    }

    // ---- epilogues ----
    if (EPI == 0 || EPI == 1) {
#pragma unroll
        for (int ms = 0; ms < 4; ++ms)
#pragma unroll
            for (int n = 0; n < 2; ++n) {
                int ng = n0 + nw * 32 + n * 16 + r;
                float bv = bias[ng];
#pragma unroll
                for (int v = 0; v < 8; ++v) {
                    int m = m0 + mw * 64 + ms * 16 + v + 8 * half;
                    float val = acc[ms][n][v] + bv;
                    if (EPI == 0)
                        outB[(size_t)m * Ncols + ng] = (bf16_t)fmaxf(val, 0.f);
                    else
                        outF[(size_t)m * Ncols + ng] = val;
                }
            }
    } else {
        // packed = (ordered(sim) << 32) | (0xFFFFFFFF - col): max() picks the
        // largest sim; ties resolve to the smallest column (jnp.argmax).
#pragma unroll
        for (int ms = 0; ms < 4; ++ms)
#pragma unroll
            for (int v = 0; v < 8; ++v) {
                unsigned long long pk = 0ull;
#pragma unroll
                for (int n = 0; n < 2; ++n) {
                    unsigned ng = (unsigned)(n0 + nw * 32 + n * 16 + r);
                    unsigned long long cand =
                        ((unsigned long long)ordered_bits(acc[ms][n][v]) << 32) |
                        (unsigned long long)(0xFFFFFFFFu - ng);
                    pk = (cand > pk) ? cand : pk;
                }
                for (int off = 1; off < 16; off <<= 1) {
                    unsigned long long o = shflxor_u64(pk, off);
                    pk = (o > pk) ? o : pk;
                }
                if (r == 0) {
                    int m = m0 + mw * 64 + ms * 16 + v + 8 * half;
                    atomicMax(&packed[m], pk);
                }
            }
    }
}

// ---------------------------------------------------------------------------
// Tail: decode argmax, gather memory row, gate MLP -> alpha, projections,
// clips, write the [449] output row. One block per batch row.
// ---------------------------------------------------------------------------
__global__ void __launch_bounds__(256)
k_tail(const unsigned long long* __restrict__ packed,
       const float* __restrict__ memory, const float* __restrict__ pfc,
       const float* __restrict__ td, const float* __restrict__ ema,
       const float* __restrict__ g1, const float* __restrict__ gb1,
       const float* __restrict__ g2, const float* __restrict__ gb2,
       const float* __restrict__ Wp, const float* __restrict__ bp,
       const float* __restrict__ Wn, const float* __restrict__ bn,
       float* __restrict__ out)
{
    __shared__ __align__(16) float rd[DM];
    __shared__ float s_alpha;

    const int b = blockIdx.x, t = threadIdx.x;

    unsigned long long pk = packed[b];
    unsigned idx  = 0xFFFFFFFFu - (unsigned)(pk & 0xFFFFFFFFull);
    float    best = ordered_decode((unsigned)(pk >> 32));

    const float* mrow = memory + (size_t)idx * DM;
    for (int c = t; c < DM; c += 256) rd[c] = mrow[c];

    if (t == 0) {
        float gi0 = best, gi1 = fabsf(td[b]), gi2 = ema[0];
        float a2 = gb2[0];
#pragma unroll
        for (int j = 0; j < 16; ++j) {
            float hj = tanhf(g1[j * 3 + 0] * gi0 + g1[j * 3 + 1] * gi1 +
                             g1[j * 3 + 2] * gi2 + gb1[j]);
            a2 += g2[j] * hj;
        }
        s_alpha = tanhf(a2);
    }
    __syncthreads();

    const float alpha = s_alpha;
    float* orow = out + (size_t)b * OUTW;
    const float4* r4 = (const float4*)rd;

    for (int o = t; o < DM; o += 256) {
        const float* wrow = (o < PP) ? (Wp + (size_t)o * DM)
                                     : (Wn + (size_t)(o - PP) * DM);
        const float4* w4 = (const float4*)wrow;
        float d = 0.f;
#pragma unroll 4
        for (int c = 0; c < DM / 4; ++c) {
            float4 a = w4[c], x = r4[c];
            d += a.x * x.x + a.y * x.y + a.z * x.z + a.w * x.w;
        }
        if (o < PP) {
            orow[o] = pfc[(size_t)b * PP + o] + alpha * (d + bp[o]);
        } else {
            int q = o - PP;
            float v = d + bn[q];
            int ch = q >> 6;                        // 0: eta, 1: decay, 2: expl
            float hi = (ch == 2) ? 0.5f : 1.0f;
            orow[PP + 1 + q] = fminf(fmaxf(v, 0.1f), hi);
        }
    }
    if (t == 0) orow[PP] = alpha;
}

// ---------------------------------------------------------------------------
// Host launcher
// ---------------------------------------------------------------------------
extern "C" void kernel_launch(void* const* d_in, const int* in_sizes, int n_in,
                              void* d_out, int out_size, void* d_ws, size_t ws_size,
                              hipStream_t stream) {
    (void)in_sizes; (void)n_in; (void)out_size; (void)ws_size;

    const float* act  = (const float*)d_in[0];
    const float* pfc  = (const float*)d_in[1];
    const float* td   = (const float*)d_in[2];
    const float* ema  = (const float*)d_in[3];
    const float* prot = (const float*)d_in[4];
    const float* mem  = (const float*)d_in[5];
    const float* W1   = (const float*)d_in[6];
    const float* b1   = (const float*)d_in[7];
    const float* W2   = (const float*)d_in[8];
    const float* b2   = (const float*)d_in[9];
    const float* g1   = (const float*)d_in[10];
    const float* gb1  = (const float*)d_in[11];
    const float* g2   = (const float*)d_in[12];
    const float* gb2  = (const float*)d_in[13];
    const float* Wp   = (const float*)d_in[14];
    const float* bp   = (const float*)d_in[15];
    const float* Wn   = (const float*)d_in[16];
    const float* bn   = (const float*)d_in[17];
    float* out = (float*)d_out;

    // ---- carve workspace (~61 MB) ----
    char* w = (char*)d_ws;
    auto carve = [&](size_t bytes) -> void* {
        void* p = (void*)w;
        w += (bytes + 255) & ~(size_t)255;
        return p;
    };
    bf16_t* W1b   = (bf16_t*)carve((size_t)HID * CMB * 2);
    bf16_t* W2b   = (bf16_t*)carve((size_t)KK * HID * 2);
    bf16_t* comb  = (bf16_t*)carve((size_t)BB * CMB * 2);
    bf16_t* pnb   = (bf16_t*)carve((size_t)NN * KK * 2);
    bf16_t* Hb    = (bf16_t*)carve((size_t)BB * HID * 2);
    float*  keyv  = (float*)carve((size_t)BB * KK * 4);
    bf16_t* knb   = (bf16_t*)carve((size_t)BB * KK * 2);
    unsigned long long* packed = (unsigned long long*)carve((size_t)BB * 8);

    // ---- 1. conversions + concat + prototype normalization ----
    k_f32_to_bf16<<<(HID * CMB + 255) / 256, 256, 0, stream>>>(W1, W1b, HID * CMB);
    k_f32_to_bf16<<<(KK * HID + 255) / 256, 256, 0, stream>>>(W2, W2b, KK * HID);
    k_build_combined<<<(BB * CMB + 255) / 256, 256, 0, stream>>>(act, pfc, comb);
    k_normalize_rows<<<NN, 256, 0, stream>>>(prot, pnb, KK);
    k_zero_u64<<<(BB + 255) / 256, 256, 0, stream>>>(packed, BB);

    // ---- 2. MLP layer 1: H = relu(comb @ W1^T + b1)  [4096 x 1024] bf16 ----
    k_gemm_wmma<0><<<dim3(HID / 128, BB / 128), 256, 0, stream>>>(
        comb, W1b, b1, nullptr, Hb, nullptr, CMB, HID);

    // ---- 3. MLP layer 2: keyv = H @ W2^T + b2  [4096 x 512] f32 ----
    k_gemm_wmma<1><<<dim3(KK / 128, BB / 128), 256, 0, stream>>>(
        Hb, W2b, b2, keyv, nullptr, nullptr, HID, KK);

    // ---- 4. normalize keyv rows -> kn (bf16) ----
    k_normalize_rows<<<BB, 256, 0, stream>>>(keyv, knb, KK);

    // ---- 5. sims = kn @ pn^T, fused row max/argmax (137 GFLOP bf16 WMMA) ----
    k_gemm_wmma<2><<<dim3(NN / 128, BB / 128), 256, 0, stream>>>(
        knb, pnb, nullptr, nullptr, nullptr, packed, KK, NN);

    // ---- 6. gather + gate MLP + projections + clips -> [4096 x 449] ----
    k_tail<<<BB, 256, 0, stream>>>(packed, mem, pfc, td, ema,
                                   g1, gb1, g2, gb2, Wp, bp, Wn, bn, out);
}
